// SelfAttention_30855045054940
// MI455X (gfx1250) — compile-verified, matrix operand order checked by reading, not run on hardware
//
#include <hip/hip_runtime.h>

// ---------------------------------------------------------------------------
// CDNA5 (gfx1250) self-attention: bf16 WMMA GEMMs + flash attention.
// B=2, S=2048, H=1024, NH=16, D=64.
// ---------------------------------------------------------------------------

typedef __attribute__((ext_vector_type(16))) __bf16 v16bf;
typedef __attribute__((ext_vector_type(8)))  __bf16 v8bf;
typedef __attribute__((ext_vector_type(8)))  float  v8f;

#define Bc 2
#define Sc 2048
#define Hc 1024
#define NHc 16
#define Dc 64

__device__ __forceinline__ unsigned short f2bf(float f) {
  unsigned int u = __float_as_uint(f);
  u += 0x7FFFu + ((u >> 16) & 1u);      // round-to-nearest-even
  return (unsigned short)(u >> 16);
}

__device__ __forceinline__ v8f v8f_zero() {
  v8f z;
#pragma unroll
  for (int i = 0; i < 8; ++i) z[i] = 0.0f;
  return z;
}

// Load one 16x32 bf16 WMMA operand fragment (A layout; B is symmetric).
// `base` points at (row * rowstride + k0) in elements; per-lane chunks:
//   chunk0 @ k = (lane<16 ? 0 : 8), chunk1 @ k = 16 + (lane<16 ? 0 : 8)
__device__ __forceinline__ v16bf load_frag(const unsigned short* base, int khalf) {
  v8bf c0 = *(const v8bf*)(base + khalf);
  v8bf c1 = *(const v8bf*)(base + 16 + khalf);
  v16bf r;
#pragma unroll
  for (int i = 0; i < 8; ++i) { r[i] = c0[i]; r[i + 8] = c1[i]; }
  return r;
}

__device__ __forceinline__ v8f wmma_bf16(v16bf a, v16bf b, v8f c) {
  return __builtin_amdgcn_wmma_f32_16x16x32_bf16(
      false, a, false, b, (short)0, c, false, false);
}

// ---------------------------------------------------------------------------
// fp32 -> bf16 bulk convert (8 elements / thread), n % 8 == 0
// ---------------------------------------------------------------------------
__global__ void cvt_f32_bf16(const float* __restrict__ src,
                             unsigned short* __restrict__ dst, int n) {
  int i = (blockIdx.x * blockDim.x + threadIdx.x) * 8;
  if (i >= n) return;
  float4 a = *(const float4*)(src + i);
  float4 b = *(const float4*)(src + i + 4);
  uint4 o;
  o.x = (unsigned)f2bf(a.x) | ((unsigned)f2bf(a.y) << 16);
  o.y = (unsigned)f2bf(a.z) | ((unsigned)f2bf(a.w) << 16);
  o.z = (unsigned)f2bf(b.x) | ((unsigned)f2bf(b.y) << 16);
  o.w = (unsigned)f2bf(b.z) | ((unsigned)f2bf(b.w) << 16);
  *(uint4*)(dst + i) = o;
}

// ---------------------------------------------------------------------------
// C[M,N] = A[M,K] * W[N,K]^T   (A,W bf16 row-major K-inner; C fp32)
// block = 128 threads = 4 waves, block tile 128x128, wave tile 64x64.
// ---------------------------------------------------------------------------
__global__ __launch_bounds__(128) void gemm_bf16_nt(
    const unsigned short* __restrict__ A, const unsigned short* __restrict__ W,
    float* __restrict__ C, int M, int N, int K) {
  const int lane  = threadIdx.x & 31;
  const int w     = threadIdx.x >> 5;
  const int mbase = blockIdx.x * 128 + (w >> 1) * 64;
  const int nbase = blockIdx.y * 128 + (w & 1) * 64;
  const int khalf = (lane < 16) ? 0 : 8;
  const int r     = lane & 15;

  v8f acc[4][4];
#pragma unroll
  for (int mf = 0; mf < 4; ++mf)
#pragma unroll
    for (int nf = 0; nf < 4; ++nf) acc[mf][nf] = v8f_zero();

  for (int kk = 0; kk < K; kk += 32) {
    v16bf af[4], bw[4];
#pragma unroll
    for (int mf = 0; mf < 4; ++mf)
      af[mf] = load_frag(A + (size_t)(mbase + mf * 16 + r) * K + kk, khalf);
#pragma unroll
    for (int nf = 0; nf < 4; ++nf)
      bw[nf] = load_frag(W + (size_t)(nbase + nf * 16 + r) * K + kk, khalf);
#pragma unroll
    for (int mf = 0; mf < 4; ++mf)
#pragma unroll
      for (int nf = 0; nf < 4; ++nf)
        acc[mf][nf] = wmma_bf16(af[mf], bw[nf], acc[mf][nf]);
  }

  const int halfoff = (lane < 16) ? 0 : 8;
#pragma unroll
  for (int mf = 0; mf < 4; ++mf)
#pragma unroll
    for (int nf = 0; nf < 4; ++nf) {
      int col = nbase + nf * 16 + r;
#pragma unroll
      for (int j = 0; j < 8; ++j) {
        int row = mbase + mf * 16 + j + halfoff;
        C[(size_t)row * N + col] = acc[mf][nf][j];
      }
    }
}

// ---------------------------------------------------------------------------
// RMSNorm + RoPE on q,k; lambda-mix on v. One wave per (b,s,h) row.
// qkv fp32 [B,S,3H]; outputs: Qo,Ko bf16 [B,NH,S,D]; Vt bf16 [B,NH,D,S].
// q is pre-scaled by 1/sqrt(D) = 0.125.
// ---------------------------------------------------------------------------
__global__ __launch_bounds__(256) void qkv_prep(
    const float* __restrict__ qkv, const float* __restrict__ ve,
    const float* __restrict__ lam, unsigned short* __restrict__ Qo,
    unsigned short* __restrict__ Ko, unsigned short* __restrict__ Vt) {
  int g    = blockIdx.x * 8 + (threadIdx.x >> 5);
  int lane = threadIdx.x & 31;
  int h = g & (NHc - 1);
  int s = (g >> 4) & (Sc - 1);
  int b = g >> 15;                                  // / (S*NH) = 2^15

  const float* rowp = qkv + (size_t)(b * Sc + s) * (3 * Hc);

  // rotate-half RoPE factors: pair (lane, lane+32), freq index = lane
  float inv_freq = exp2f(-(float)lane * (13.2877123795494f / 32.0f)); // 10000^(-lane/32)
  float ang = (float)s * inv_freq;
  float sn, cs;
  __sincosf(ang, &sn, &cs);

  size_t outrow = ((size_t)(b * NHc + h) * Sc + s) * Dc;

  // ---- q ----
  {
    float x1 = rowp[h * Dc + lane];
    float x2 = rowp[h * Dc + lane + 32];
    float ss = x1 * x1 + x2 * x2;
    ss += __shfl_xor(ss, 1, 32);  ss += __shfl_xor(ss, 2, 32);
    ss += __shfl_xor(ss, 4, 32);  ss += __shfl_xor(ss, 8, 32);
    ss += __shfl_xor(ss, 16, 32);
    float rn = rsqrtf(ss * (1.0f / 64.0f) + 1.1920929e-07f);
    x1 *= rn; x2 *= rn;
    Qo[outrow + lane]      = f2bf((x1 * cs + x2 * sn) * 0.125f);
    Qo[outrow + lane + 32] = f2bf((-x1 * sn + x2 * cs) * 0.125f);
  }
  // ---- k ----
  {
    float x1 = rowp[Hc + h * Dc + lane];
    float x2 = rowp[Hc + h * Dc + lane + 32];
    float ss = x1 * x1 + x2 * x2;
    ss += __shfl_xor(ss, 1, 32);  ss += __shfl_xor(ss, 2, 32);
    ss += __shfl_xor(ss, 4, 32);  ss += __shfl_xor(ss, 8, 32);
    ss += __shfl_xor(ss, 16, 32);
    float rn = rsqrtf(ss * (1.0f / 64.0f) + 1.1920929e-07f);
    x1 *= rn; x2 *= rn;
    Ko[outrow + lane]      = f2bf(x1 * cs + x2 * sn);
    Ko[outrow + lane + 32] = f2bf(-x1 * sn + x2 * cs);
  }
  // ---- v (transposed store: [B,NH,D,S]) ----
  {
    float l0 = lam[0], l1 = lam[1];
    const float* vep = ve + (size_t)(b * Sc + s) * Hc + h * Dc;
    float v1 = l0 * rowp[2 * Hc + h * Dc + lane]      + l1 * vep[lane];
    float v2 = l0 * rowp[2 * Hc + h * Dc + lane + 32] + l1 * vep[lane + 32];
    size_t vtbase = (size_t)(b * NHc + h) * Dc;
    Vt[(vtbase + lane) * Sc + s]      = f2bf(v1);
    Vt[(vtbase + lane + 32) * Sc + s] = f2bf(v2);
  }
}

// ---------------------------------------------------------------------------
// Causal flash attention. One wave handles a 16-row Q tile for one (b,h),
// streaming 64-wide K/V tiles with online softmax. WMMA for QK^T and PV;
// P goes through LDS to convert C-layout -> A-layout.
// Q,K: bf16 [B,NH,S,D]; Vt: bf16 [B,NH,D,S]; O: bf16 [B,S,H].
// ---------------------------------------------------------------------------
__global__ __launch_bounds__(128) void flash_attn(
    const unsigned short* __restrict__ Q, const unsigned short* __restrict__ Kb,
    const unsigned short* __restrict__ Vt, unsigned short* __restrict__ O) {
  __shared__ __align__(16) unsigned short pls[4][16][64];

  const int lane  = threadIdx.x & 31;
  const int w     = threadIdx.x >> 5;
  const int h     = blockIdx.y;
  const int b     = blockIdx.z;
  const int qbase = (blockIdx.x * 4 + w) * 16;
  const int khalf = (lane < 16) ? 0 : 8;
  const int r     = lane & 15;
  const int rowoff = (lane < 16) ? 0 : 8;   // C-layout row offset per lane half

  const size_t head = (size_t)(b * NHc + h) * Sc * Dc;
  const unsigned short* qp = Q  + head;
  const unsigned short* kp = Kb + head;
  const unsigned short* vp = Vt + head;     // per-head [D][S]

  // Q A-operand fragments (16 x 64 = two 16x32 frags), resident for the loop
  v16bf qa0 = load_frag(qp + (size_t)(qbase + r) * Dc + 0,  khalf);
  v16bf qa1 = load_frag(qp + (size_t)(qbase + r) * Dc + 32, khalf);

  v8f oacc[4];
#pragma unroll
  for (int df = 0; df < 4; ++df) oacc[df] = v8f_zero();
  float m[8], lsum[8];
#pragma unroll
  for (int j = 0; j < 8; ++j) { m[j] = -__builtin_inff(); lsum[j] = 0.0f; }

  const int lastKt = qbase >> 6;
  for (int kt = 0; kt <= lastKt; ++kt) {
    const int kbase = kt * 64;

    // ---- scores: S = Q K^T  (16 x 64 tile = 4 N-fragments) ----
    v8f sc[4];
#pragma unroll
    for (int nf = 0; nf < 4; ++nf) {
      v16bf kb0 = load_frag(kp + (size_t)(kbase + nf * 16 + r) * Dc + 0,  khalf);
      v16bf kb1 = load_frag(kp + (size_t)(kbase + nf * 16 + r) * Dc + 32, khalf);
      v8f s = v8f_zero();
      s = wmma_bf16(qa0, kb0, s);
      s = wmma_bf16(qa1, kb1, s);
      sc[nf] = s;
    }

    // ---- causal mask on the diagonal tile ----
    if (kt == lastKt) {
#pragma unroll
      for (int nf = 0; nf < 4; ++nf) {
        int col = kbase + nf * 16 + r;
#pragma unroll
        for (int j = 0; j < 8; ++j) {
          int row = qbase + j + rowoff;
          if (col > row) sc[nf][j] = -1.0e30f;
        }
      }
    }

    // ---- online softmax (row stats live replicated across each 16-lane half)
    float scl[8];
#pragma unroll
    for (int j = 0; j < 8; ++j) {
      float v = fmaxf(fmaxf(sc[0][j], sc[1][j]), fmaxf(sc[2][j], sc[3][j]));
      v = fmaxf(v, __shfl_xor(v, 1, 32));
      v = fmaxf(v, __shfl_xor(v, 2, 32));
      v = fmaxf(v, __shfl_xor(v, 4, 32));
      v = fmaxf(v, __shfl_xor(v, 8, 32));
      float mn = fmaxf(m[j], v);
      scl[j] = __expf(m[j] - mn);
      m[j] = mn;
    }
#pragma unroll
    for (int nf = 0; nf < 4; ++nf)
#pragma unroll
      for (int j = 0; j < 8; ++j) sc[nf][j] = __expf(sc[nf][j] - m[j]);
#pragma unroll
    for (int j = 0; j < 8; ++j) {
      float t = sc[0][j] + sc[1][j] + sc[2][j] + sc[3][j];
      t += __shfl_xor(t, 1, 32);
      t += __shfl_xor(t, 2, 32);
      t += __shfl_xor(t, 4, 32);
      t += __shfl_xor(t, 8, 32);
      lsum[j] = lsum[j] * scl[j] + t;
    }
#pragma unroll
    for (int df = 0; df < 4; ++df)
#pragma unroll
      for (int j = 0; j < 8; ++j) oacc[df][j] *= scl[j];

    // ---- P: C-layout regs -> LDS (row major bf16) -> A-layout frags ----
#pragma unroll
    for (int nf = 0; nf < 4; ++nf)
#pragma unroll
      for (int j = 0; j < 8; ++j)
        pls[w][j + rowoff][nf * 16 + r] = f2bf(sc[nf][j]);

    v16bf pa0 = load_frag(&pls[w][r][0],  khalf);
    v16bf pa1 = load_frag(&pls[w][r][32], khalf);

    // ---- O += P V  (V^T stored [D][S] so B-operand loads are contiguous) ----
#pragma unroll
    for (int df = 0; df < 4; ++df) {
      v16bf vb0 = load_frag(vp + (size_t)(df * 16 + r) * Sc + kbase + 0,  khalf);
      v16bf vb1 = load_frag(vp + (size_t)(df * 16 + r) * Sc + kbase + 32, khalf);
      oacc[df] = wmma_bf16(pa0, vb0, oacc[df]);
      oacc[df] = wmma_bf16(pa1, vb1, oacc[df]);
    }
  }

  // ---- normalize and write attn output bf16 [B,S,H], col = h*64 + d ----
  float inv[8];
#pragma unroll
  for (int j = 0; j < 8; ++j) inv[j] = 1.0f / lsum[j];
#pragma unroll
  for (int df = 0; df < 4; ++df) {
    int col = h * Dc + df * 16 + r;
#pragma unroll
    for (int j = 0; j < 8; ++j) {
      int row = qbase + j + rowoff;
      O[(size_t)(b * Sc + row) * Hc + col] = f2bf(oacc[df][j] * inv[j]);
    }
  }
}

// ---------------------------------------------------------------------------
// host launcher
// ---------------------------------------------------------------------------
extern "C" void kernel_launch(void* const* d_in, const int* in_sizes, int n_in,
                              void* d_out, int out_size, void* d_ws, size_t ws_size,
                              hipStream_t stream) {
  (void)in_sizes; (void)n_in; (void)out_size; (void)ws_size;
  const float* x    = (const float*)d_in[0];   // [B,S,H]
  const float* ve   = (const float*)d_in[1];   // [B,S,H]
  const float* Wqkv = (const float*)d_in[2];   // [3H,H]
  const float* Wo   = (const float*)d_in[3];   // [H,H]
  const float* lam  = (const float*)d_in[4];   // [2]
  float* out = (float*)d_out;                  // [B,S,H] fp32

  const size_t MB = 1ull << 20;
  char* ws = (char*)d_ws;
  unsigned short* x_bf    = (unsigned short*)(ws + 0 * MB);   // 8 MB
  unsigned short* wqkv_bf = (unsigned short*)(ws + 8 * MB);   // 6 MB
  unsigned short* wo_bf   = (unsigned short*)(ws + 14 * MB);  // 2 MB
  float*          qkv     = (float*)(ws + 16 * MB);           // 48 MB
  unsigned short* q_bf    = (unsigned short*)(ws + 64 * MB);  // 8 MB
  unsigned short* k_bf    = (unsigned short*)(ws + 72 * MB);  // 8 MB
  unsigned short* v_t     = (unsigned short*)(ws + 80 * MB);  // 8 MB
  unsigned short* attn    = (unsigned short*)(ws + 88 * MB);  // 8 MB (96 MB total)

  // 1) fp32 -> bf16 operand conversion
  cvt_f32_bf16<<<2048, 256, 0, stream>>>(x,    x_bf,    Bc * Sc * Hc);
  cvt_f32_bf16<<<1536, 256, 0, stream>>>(Wqkv, wqkv_bf, 3 * Hc * Hc);
  cvt_f32_bf16<<<512,  256, 0, stream>>>(Wo,   wo_bf,   Hc * Hc);

  // 2) qkv = x @ Wqkv^T   (M=4096, N=3072, K=1024)
  gemm_bf16_nt<<<dim3(32, 24), 128, 0, stream>>>(x_bf, wqkv_bf, qkv,
                                                 Bc * Sc, 3 * Hc, Hc);

  // 3) rmsnorm + rope + lambda-mix, layout change
  qkv_prep<<<8192, 256, 0, stream>>>(qkv, ve, lam, q_bf, k_bf, v_t);

  // 4) causal flash attention
  flash_attn<<<dim3(Sc / 64, NHc, Bc), 128, 0, stream>>>(q_bf, k_bf, v_t, attn);

  // 5) out = attn @ Wo^T   (M=4096, N=1024, K=1024)
  gemm_bf16_nt<<<dim3(32, 8), 128, 0, stream>>>(attn, wo_bf, out,
                                                Bc * Sc, Hc, Hc);
}
